// EDQALite_39067022524477
// MI455X (gfx1250) — matrix operand order repaired and checked by reading, not compiled, not run on hardware
//
#include <hip/hip_runtime.h>
#include <hip/hip_bf16.h>

typedef __attribute__((ext_vector_type(16))) _Float16 v16h;
typedef __attribute__((ext_vector_type(8)))  float    v8f;

#define C_    256
#define HW_   16384L
#define SRD   264      // LDS half stride for [64][256] f16 tiles
#define PST   66       // softmax P tile stride (halves)
#define OST   260      // f32 output staging stride
#define WSTR  40       // staged weight slice row stride (halves): 80B rows, conflict-free
#define WBUFH 10240    // one weight slice buffer: 256 rows * 40 halves

#define WMMAF16(a, b, c) \
  __builtin_amdgcn_wmma_f32_16x16x32_f16(false, (a), false, (b), (short)0, (c), false, false)

// ---- CDNA5 async global->LDS (ASYNCcnt) ----
__device__ __forceinline__ void async_ld_b128(_Float16* lds_dst, const _Float16* gsrc) {
  const unsigned off = (unsigned)(unsigned long long)lds_dst;   // low 32 bits = LDS byte offset
  asm volatile("global_load_async_to_lds_b128 %0, %1, off" :: "v"(off), "v"(gsrc) : "memory");
}
__device__ __forceinline__ void async_wait0() {
  asm volatile("s_wait_asynccnt 0x0" ::: "memory");
}

// Stage one 256x32 f16 weight k-slice (16KB) into LDS, cooperative across 256 threads.
// Source: w[n*wstride + k0 + k]; dest: dst[n*WSTR + k]. 4 async b128 per thread.
__device__ __forceinline__ void stage_wslice(const _Float16* __restrict__ w, int wstride, int k0,
                                             _Float16* dst, int tid) {
  const int chunk = (tid & 3) * 8;   // halves within row
  const int rbase = tid >> 2;        // 0..63
#pragma unroll
  for (int p = 0; p < 4; ++p) {
    const int n = rbase + p * 64;
    async_ld_b128(dst + n * WSTR + chunk, w + n * wstride + k0 + chunk);
  }
}

// ---- WMMA fragment loaders (wave32; k = 16*(lane/16) + j, same order in A and B) ----
__device__ __forceinline__ v16h frag_A(const _Float16* x, int stride, int m0, int k0, int lane) {
  const int m  = m0 + (lane & 15);
  const int kb = k0 + ((lane >> 4) << 4);
  const _Float16* p = x + m * stride + kb;
  v16h a;
#pragma unroll
  for (int j = 0; j < 16; ++j) a[j] = p[j];
  return a;
}
__device__ __forceinline__ v16h frag_B(const _Float16* w, int stride, int n0, int k0, int lane) {
  const int n  = n0 + (lane & 15);
  const int kb = k0 + ((lane >> 4) << 4);
  const _Float16* p = w + n * stride + kb;
  v16h b;
#pragma unroll
  for (int j = 0; j < 16; ++j) b[j] = p[j];
  return b;
}
__device__ __forceinline__ v16h frag_B_kmaj(const _Float16* w, int stride, int n0, int k0, int lane) {
  const int n  = n0 + (lane & 15);
  const int kb = k0 + ((lane >> 4) << 4);
  v16h b;
#pragma unroll
  for (int j = 0; j < 16; ++j) b[j] = w[(kb + j) * stride + n];
  return b;
}

// GroupNorm affine + exact GELU, applied on the fly to guide loads
__device__ __forceinline__ float gn_gelu_apply(float x, float m, float r, float g, float b) {
  const float y = (x - m) * r * g + b;
  return 0.5f * y * (1.f + erff(y * 0.70710678118654752f));
}

// ================= K0a: weight conversion + LN-affine folding =================
__global__ __launch_bounds__(256) void prep_w_kernel(
    const float* __restrict__ gw, const float* __restrict__ ipw,
    const float* __restrict__ qn_g, const float* __restrict__ kn_g, const float* __restrict__ vn_g,
    const float* __restrict__ aow, const float* __restrict__ opw,
    _Float16* __restrict__ gwh, _Float16* __restrict__ wq, _Float16* __restrict__ wk,
    _Float16* __restrict__ wv, _Float16* __restrict__ aowh, _Float16* __restrict__ opwh)
{
  const int i = blockIdx.x * 256 + threadIdx.x;
  if (i < 196608) gwh[i] = (_Float16)gw[i];
  if (i < 65536) {
    const int c = i & 255;
    wq[i]   = (_Float16)(ipw[i]           * qn_g[c]);
    wk[i]   = (_Float16)(ipw[65536 + i]   * kn_g[c]);
    wv[i]   = (_Float16)(ipw[131072 + i]  * vn_g[c]);
    aowh[i] = (_Float16)aow[i];
    opwh[i] = (_Float16)opw[i];
  }
}

// ================= K0b: effective biases (fold LN beta through W) =================
__global__ void prep_bias_kernel(
    const float* __restrict__ ipw, const float* __restrict__ ipb,
    const float* __restrict__ qn_b, const float* __restrict__ kn_b, const float* __restrict__ vn_b,
    float* __restrict__ bq, float* __restrict__ bk, float* __restrict__ bv)
{
  const int o = threadIdx.x;  // 256
  float sq = 0.f, sk = 0.f, sv = 0.f;
  for (int c = 0; c < 256; ++c) {
    sq += qn_b[c] * ipw[o * 256 + c];
    sk += kn_b[c] * ipw[(256 + o) * 256 + c];
    sv += vn_b[c] * ipw[(512 + o) * 256 + c];
  }
  bq[o] = ipb[o] + sq;
  bk[o] = ipb[256 + o] + sk;
  bv[o] = ipb[512 + o] + sv;
}

// ================= K1: guide GEMM with async-staged weights =================
__global__ __launch_bounds__(256) void guide_gemm_kernel(
    const float* __restrict__ pre, const float* __restrict__ post,
    const float* __restrict__ fused, const _Float16* __restrict__ gwh,
    float* __restrict__ gpre)
{
  extern __shared__ char smem[];
  _Float16* A    = (_Float16*)smem;              // [64][776] f16 = 99,328 B
  float*    outb = (float*)smem;                 // reused after GEMM: [64][OST] f32
  _Float16* wst  = (_Float16*)(smem + 99328);    // 2 x WBUFH halves (40,960 B)
  const int tid = threadIdx.x;
  const int b   = blockIdx.x >> 8;
  const int blk = blockIdx.x & 255;
  const long pix0 = (long)blk * 64;
  const long cb   = (long)b * C_ * HW_;
  const int t = tid & 63, qq = tid >> 6;

  stage_wslice(gwh, 768, 0, wst, tid);   // prefetch k-slice 0 under the A fill

  for (int i = 0; i < 192; ++i) {
    const int c3 = qq * 192 + i;
    float x;
    if (c3 < 256)      x = pre  [cb + (long)c3 * HW_ + pix0 + t];
    else if (c3 < 512) x = post [cb + (long)(c3 - 256) * HW_ + pix0 + t];
    else               x = fused[cb + (long)(c3 - 512) * HW_ + pix0 + t];
    A[t * 776 + c3] = (_Float16)x;
  }
  async_wait0();
  __syncthreads();

  const int wv_id = tid >> 5, lane = tid & 31;
  const int mt = wv_id & 3, nt0 = (wv_id >> 2) * 8;
  v8f acc[8] = {};
  for (int kk = 0; kk < 24; ++kk) {
    const _Float16* cur = wst + (kk & 1) * WBUFH;
    if (kk + 1 < 24) stage_wslice(gwh, 768, (kk + 1) * 32, wst + ((kk + 1) & 1) * WBUFH, tid);
    const v16h a = frag_A(A, 776, mt * 16, kk * 32, lane);
#pragma unroll
    for (int j = 0; j < 8; ++j) {
      const v16h bf = frag_B(cur, WSTR, (nt0 + j) * 16, 0, lane);
      acc[j] = WMMAF16(a, bf, acc[j]);
    }
    async_wait0();
    __syncthreads();
  }
#pragma unroll
  for (int j = 0; j < 8; ++j) {
    const int o = (nt0 + j) * 16 + (lane & 15);
    const int r0 = (lane >> 4) * 8;
#pragma unroll
    for (int r = 0; r < 8; ++r) outb[(mt * 16 + r0 + r) * OST + o] = acc[j][r];
  }
  __syncthreads();
  for (int jo = 0; jo < 64; ++jo) {
    const int o = jo * 4 + qq;
    gpre[cb + (long)o * HW_ + pix0 + t] = outb[t * OST + o];
  }
}

// ================= K2: group-norm statistics (deterministic) =================
__global__ __launch_bounds__(256) void gstats_kernel(
    const float* __restrict__ gpre, float* __restrict__ mu, float* __restrict__ rs)
{
  const int bg = blockIdx.x;  // 64 = B*GROUPS
  const int b = bg >> 3, g = bg & 7;
  const float* base = gpre + ((long)b * C_ + (long)g * 32) * HW_;
  float s = 0.f, ss = 0.f;
  for (long i = threadIdx.x; i < 32 * HW_; i += 256) {
    const float x = base[i];
    s += x; ss += x * x;
  }
  __shared__ float r0[256], r1[256];
  r0[threadIdx.x] = s; r1[threadIdx.x] = ss;
  __syncthreads();
  for (int k = 128; k > 0; k >>= 1) {
    if (threadIdx.x < (unsigned)k) { r0[threadIdx.x] += r0[threadIdx.x + k]; r1[threadIdx.x] += r1[threadIdx.x + k]; }
    __syncthreads();
  }
  if (threadIdx.x == 0) {
    const float m = r0[0] * (1.f / 524288.f);
    const float v = r1[0] * (1.f / 524288.f) - m * m;
    mu[bg] = m;
    rs[bg] = rsqrtf(v + 1e-5f);
  }
}

// ========== K4: fused window attention (GN+GELU folded into guide loads) ==========
__global__ __launch_bounds__(256) void win_attn_kernel(
    const float* __restrict__ dmg, const float* __restrict__ gpre,
    const float* __restrict__ gmu, const float* __restrict__ grs,
    const float* __restrict__ gn_g, const float* __restrict__ gn_b,
    const _Float16* __restrict__ wq, const float* __restrict__ bq,
    const _Float16* __restrict__ wk, const float* __restrict__ bk,
    const _Float16* __restrict__ wv, const float* __restrict__ bv,
    const _Float16* __restrict__ wo, const float* __restrict__ bo,
    float* __restrict__ calib)
{
  extern __shared__ char smem[];
  _Float16* xh_d = (_Float16*)smem;          // 64*264 f16 each
  _Float16* xh_g = xh_d + 64 * SRD;
  _Float16* qs   = xh_g + 64 * SRD;
  _Float16* ksm  = qs   + 64 * SRD;
  _Float16* vs   = ksm  + 64 * SRD;
  _Float16* cs   = vs   + 64 * SRD;
  _Float16* Pb   = cs   + 64 * SRD;          // 8 * 16*66 f16
  float*    red  = (float*)(Pb + 8 * 16 * PST);   // 1280 f32
  _Float16* wst  = (_Float16*)(smem + 224768);    // 2 x WBUFH halves
  float*    outb = (float*)smem;             // reuse xh_d+xh_g: [64][OST] f32

  const int tid = threadIdx.x;
  const int b   = blockIdx.x >> 8;
  const int wy  = (blockIdx.x >> 4) & 15;
  const int wx  = blockIdx.x & 15;
  const int h0  = wy * 8, w0 = wx * 8;

  const int t  = tid & 63;
  const int qq = tid >> 6;
  const long pixb = (long)(h0 + (t >> 3)) * 128 + (w0 + (t & 7));
  const long cb   = (long)b * C_ * HW_;

  stage_wslice(wq, 256, 0, wst, tid);    // prefetch first QKV weight slice under LN phase

  // ---- per-token LayerNorm stats over C (dmg raw; guide = gelu(gn(gpre))) ----
  float sd = 0.f, ssd = 0.f, sg = 0.f, ssg = 0.f;
  for (int i = 0; i < 64; ++i) {
    const int c = qq * 64 + i;
    const int bg = b * 8 + (c >> 5);
    const float xd = dmg[cb + (long)c * HW_ + pixb];
    const float xg = gn_gelu_apply(gpre[cb + (long)c * HW_ + pixb],
                                   gmu[bg], grs[bg], gn_g[c], gn_b[c]);
    sd += xd; ssd += xd * xd; sg += xg; ssg += xg * xg;
  }
  red[t * 4 + qq]       = sd;
  red[256 + t * 4 + qq] = ssd;
  red[512 + t * 4 + qq] = sg;
  red[768 + t * 4 + qq] = ssg;
  __syncthreads();
  if (tid < 64) {
    float a0 = 0.f, a1 = 0.f, a2 = 0.f, a3 = 0.f;
    for (int j = 0; j < 4; ++j) {
      a0 += red[tid * 4 + j];       a1 += red[256 + tid * 4 + j];
      a2 += red[512 + tid * 4 + j]; a3 += red[768 + tid * 4 + j];
    }
    const float md = a0 * (1.f / 256.f), mg = a2 * (1.f / 256.f);
    red[1024 + tid] = md;
    red[1088 + tid] = rsqrtf(a1 * (1.f / 256.f) - md * md + 1e-5f);
    red[1152 + tid] = mg;
    red[1216 + tid] = rsqrtf(a3 * (1.f / 256.f) - mg * mg + 1e-5f);
  }
  __syncthreads();
  const float mud = red[1024 + t], rsd = red[1088 + t];
  const float mug = red[1152 + t], rsg = red[1216 + t];
  for (int i = 0; i < 64; ++i) {
    const int c = qq * 64 + i;
    const int bg = b * 8 + (c >> 5);
    const float xd = dmg[cb + (long)c * HW_ + pixb];
    const float xg = gn_gelu_apply(gpre[cb + (long)c * HW_ + pixb],
                                   gmu[bg], grs[bg], gn_g[c], gn_b[c]);
    xh_d[t * SRD + c] = (_Float16)((xd - mud) * rsd);
    xh_g[t * SRD + c] = (_Float16)((xg - mug) * rsg);
  }
  async_wait0();
  __syncthreads();

  const int wv_id = tid >> 5, lane = tid & 31;
  const int mt = wv_id & 3, nt0 = (wv_id >> 2) * 8;
  const v8f vzero = {};

  // ---- QKV projections: 24-step pipeline with double-buffered weight slices ----
  {
    const _Float16* srcs[3] = { xh_d, xh_d, xh_g };
    const _Float16* wts[3]  = { wq, wk, wv };
    const float*    bss[3]  = { bq, bk, bv };
    _Float16*       dsts[3] = { qs, ksm, vs };
    v8f acc[8];
#pragma unroll
    for (int j = 0; j < 8; ++j) acc[j] = vzero;
    for (int s = 0; s < 24; ++s) {
      const int m3 = s >> 3, kk = s & 7;
      const _Float16* cur = wst + (s & 1) * WBUFH;
      if (s + 1 < 24)
        stage_wslice(wts[(s + 1) >> 3], 256, ((s + 1) & 7) * 32,
                     wst + ((s + 1) & 1) * WBUFH, tid);
      const v16h a = frag_A(srcs[m3], SRD, mt * 16, kk * 32, lane);
#pragma unroll
      for (int j = 0; j < 8; ++j) {
        const v16h bf = frag_B(cur, WSTR, (nt0 + j) * 16, 0, lane);
        acc[j] = WMMAF16(a, bf, acc[j]);
      }
      if (kk == 7) {  // matrix finished: epilogue + reset accumulators
#pragma unroll
        for (int j = 0; j < 8; ++j) {
          const int o = (nt0 + j) * 16 + (lane & 15);
          const float bias = bss[m3][o];
          const int r0 = (lane >> 4) * 8;
#pragma unroll
          for (int r = 0; r < 8; ++r)
            dsts[m3][(mt * 16 + r0 + r) * SRD + o] = (_Float16)(acc[j][r] + bias);
          acc[j] = vzero;
        }
      }
      async_wait0();
      __syncthreads();
    }
  }

  // ---- attention: wave handles head h ----
  {
    const int h = wv_id;
    _Float16* Pw = Pb + h * 16 * PST;
    const float scale = 0.17677669529663687f;  // 1/sqrt(32)
    for (int m2 = 0; m2 < 4; ++m2) {
      v8f sa[4];
#pragma unroll
      for (int j = 0; j < 4; ++j) sa[j] = vzero;
      const v16h aq = frag_A(qs + h * 32, SRD, m2 * 16, 0, lane);
#pragma unroll
      for (int j = 0; j < 4; ++j) {
        const v16h bk_ = frag_B(ksm + h * 32, SRD, j * 16, 0, lane);
        sa[j] = WMMAF16(aq, bk_, sa[j]);
      }
#pragma unroll
      for (int r = 0; r < 8; ++r) {
        float mx = fmaxf(fmaxf(sa[0][r], sa[1][r]), fmaxf(sa[2][r], sa[3][r]));
        for (int o2 = 1; o2 < 16; o2 <<= 1) mx = fmaxf(mx, __shfl_xor(mx, o2, 32));
        float sum = 0.f;
#pragma unroll
        for (int j = 0; j < 4; ++j) {
          const float p = __expf((sa[j][r] - mx) * scale);
          sa[j][r] = p; sum += p;
        }
        for (int o2 = 1; o2 < 16; o2 <<= 1) sum += __shfl_xor(sum, o2, 32);
        const float inv = 1.f / sum;
#pragma unroll
        for (int j = 0; j < 4; ++j) sa[j][r] *= inv;
      }
      const int r0 = (lane >> 4) * 8;
#pragma unroll
      for (int j = 0; j < 4; ++j)
#pragma unroll
        for (int r = 0; r < 8; ++r)
          Pw[(r0 + r) * PST + j * 16 + (lane & 15)] = (_Float16)sa[j][r];
      v8f ca[2];
      ca[0] = vzero; ca[1] = vzero;
      for (int kk = 0; kk < 2; ++kk) {
        const v16h ap = frag_A(Pw, PST, 0, kk * 32, lane);
#pragma unroll
        for (int j = 0; j < 2; ++j) {
          const v16h bv_ = frag_B_kmaj(vs + h * 32, SRD, j * 16, kk * 32, lane);
          ca[j] = WMMAF16(ap, bv_, ca[j]);
        }
      }
#pragma unroll
      for (int j = 0; j < 2; ++j)
#pragma unroll
        for (int r = 0; r < 8; ++r)
          cs[(m2 * 16 + r0 + r) * SRD + h * 32 + j * 16 + (lane & 15)] = (_Float16)ca[j][r];
    }
  }
  stage_wslice(wo, 256, 0, wst, tid);   // prefetch attn_out slice 0 (wst idle since QKV)
  async_wait0();
  __syncthreads();

  // ---- attn_out projection with staged weights ----
  {
    v8f acc[8];
#pragma unroll
    for (int j = 0; j < 8; ++j) acc[j] = vzero;
    for (int kk = 0; kk < 8; ++kk) {
      const _Float16* cur = wst + (kk & 1) * WBUFH;
      if (kk + 1 < 8) stage_wslice(wo, 256, (kk + 1) * 32, wst + ((kk + 1) & 1) * WBUFH, tid);
      const v16h a = frag_A(cs, SRD, mt * 16, kk * 32, lane);
#pragma unroll
      for (int j = 0; j < 8; ++j) {
        const v16h bf = frag_B(cur, WSTR, (nt0 + j) * 16, 0, lane);
        acc[j] = WMMAF16(a, bf, acc[j]);
      }
      async_wait0();
      __syncthreads();
    }
#pragma unroll
    for (int j = 0; j < 8; ++j) {
      const int o = (nt0 + j) * 16 + (lane & 15);
      const float bias = bo[o];
      const int r0 = (lane >> 4) * 8;
#pragma unroll
      for (int r = 0; r < 8; ++r)
        outb[(mt * 16 + r0 + r) * OST + o] = acc[j][r] + bias;
    }
  }
  __syncthreads();

  for (int jo = 0; jo < 64; ++jo) {
    const int o = jo * 4 + qq;
    calib[cb + (long)o * HW_ + pixb] = outb[t * OST + o];
  }
}

// ===== K5: out_proj GEMM + residual + per-WG norm partials (staged weights) =====
__global__ __launch_bounds__(256) void outproj_kernel(
    const float* __restrict__ calib, const float* __restrict__ fused,
    const _Float16* __restrict__ opwh, const float* __restrict__ alpha_p,
    float* __restrict__ refined, float* __restrict__ partials)
{
  extern __shared__ char smem[];
  _Float16* A    = (_Float16*)smem;             // [64][264] f16
  float*    outb = (float*)smem;                // reuse: [64][OST] f32
  float*    red  = (float*)(smem + 66560);      // 256 f32
  _Float16* wst  = (_Float16*)(smem + 67584);   // 2 x WBUFH halves
  const int tid = threadIdx.x;
  const int b   = blockIdx.x >> 8;
  const int blk = blockIdx.x & 255;
  const long pix0 = (long)blk * 64;
  const long cb   = (long)b * C_ * HW_;
  const int t = tid & 63, qq = tid >> 6;

  stage_wslice(opwh, 256, 0, wst, tid);

  float ssq = 0.f;
  for (int i = 0; i < 64; ++i) {
    const int c = qq * 64 + i;
    const float x = calib[cb + (long)c * HW_ + pix0 + t];
    ssq += x * x;
    A[t * SRD + c] = (_Float16)x;
  }
  async_wait0();
  red[tid] = ssq;
  __syncthreads();
  for (int s2 = 128; s2 > 0; s2 >>= 1) {
    if (tid < s2) red[tid] += red[tid + s2];
    __syncthreads();
  }
  if (tid == 0) partials[blockIdx.x] = red[0];

  const int wv_id = tid >> 5, lane = tid & 31;
  const int mt = wv_id & 3, nt0 = (wv_id >> 2) * 8;
  v8f acc[8] = {};
  for (int kk = 0; kk < 8; ++kk) {
    const _Float16* cur = wst + (kk & 1) * WBUFH;
    if (kk + 1 < 8) stage_wslice(opwh, 256, (kk + 1) * 32, wst + ((kk + 1) & 1) * WBUFH, tid);
    const v16h a = frag_A(A, SRD, mt * 16, kk * 32, lane);
#pragma unroll
    for (int j = 0; j < 8; ++j) {
      const v16h bf = frag_B(cur, WSTR, (nt0 + j) * 16, 0, lane);
      acc[j] = WMMAF16(a, bf, acc[j]);
    }
    async_wait0();
    __syncthreads();
  }
#pragma unroll
  for (int j = 0; j < 8; ++j) {
    const int o = (nt0 + j) * 16 + (lane & 15);
    const int r0 = (lane >> 4) * 8;
#pragma unroll
    for (int r = 0; r < 8; ++r)
      outb[(mt * 16 + r0 + r) * OST + o] = acc[j][r];
  }
  __syncthreads();
  const float alpha = alpha_p[0];
  for (int jo = 0; jo < 64; ++jo) {
    const int o = jo * 4 + qq;
    const long idx = cb + (long)o * HW_ + pix0 + t;
    refined[idx] = fused[idx] + alpha * outb[t * OST + o];
  }
}

// ================= K6: finalize norms + alpha_vec (deterministic order) =================
__global__ void finalize_kernel(const float* __restrict__ partials,
                                const float* __restrict__ alpha_p,
                                float* __restrict__ alpha_vec,
                                float* __restrict__ norms)
{
  const int b = threadIdx.x;
  if (b < 8) {
    float s = 0.f;
    for (int i = 0; i < 256; ++i) s += partials[b * 256 + i];
    norms[b] = sqrtf(s);
    alpha_vec[b] = alpha_p[0];
  }
}

extern "C" void kernel_launch(void* const* d_in, const int* in_sizes, int n_in,
                              void* d_out, int out_size, void* d_ws, size_t ws_size,
                              hipStream_t stream) {
  (void)in_sizes; (void)n_in; (void)out_size; (void)ws_size;
  const float* fused = (const float*)d_in[0];
  const float* dmg   = (const float*)d_in[1];
  const float* pre   = (const float*)d_in[2];
  const float* post  = (const float*)d_in[3];
  const float* gw    = (const float*)d_in[4];
  const float* gn_g  = (const float*)d_in[5];
  const float* gn_b  = (const float*)d_in[6];
  const float* qn_g  = (const float*)d_in[7];
  const float* qn_b  = (const float*)d_in[8];
  const float* kn_g  = (const float*)d_in[9];
  const float* kn_b  = (const float*)d_in[10];
  const float* vn_g  = (const float*)d_in[11];
  const float* vn_b  = (const float*)d_in[12];
  const float* ipw   = (const float*)d_in[13];
  const float* ipb   = (const float*)d_in[14];
  const float* aow   = (const float*)d_in[15];
  const float* aob   = (const float*)d_in[16];
  const float* opw   = (const float*)d_in[17];
  const float* alpha = (const float*)d_in[18];

  float* out       = (float*)d_out;
  float* refined   = out;                    // 33,554,432
  float* calib     = out + 33554432L;        // 33,554,432
  float* alpha_vec = out + 67108864L;        // 8
  float* norms     = out + 67108872L;        // 8

  char* ws = (char*)d_ws;
  float*    gbuf = (float*)ws;                              // 134,217,728 B (pre-GN guide)
  size_t off = 134217728UL;
  _Float16* gwh  = (_Float16*)(ws + off); off += 393216;
  _Float16* wq   = (_Float16*)(ws + off); off += 131072;
  _Float16* wk   = (_Float16*)(ws + off); off += 131072;
  _Float16* wvv  = (_Float16*)(ws + off); off += 131072;
  _Float16* aowh = (_Float16*)(ws + off); off += 131072;
  _Float16* opwh = (_Float16*)(ws + off); off += 131072;
  float* bq = (float*)(ws + off); off += 1024;
  float* bk = (float*)(ws + off); off += 1024;
  float* bv = (float*)(ws + off); off += 1024;
  float* mu = (float*)(ws + off); off += 256;
  float* rs = (float*)(ws + off); off += 256;
  float* partials = (float*)(ws + off); off += 8192;

  prep_w_kernel<<<768, 256, 0, stream>>>(gw, ipw, qn_g, kn_g, vn_g, aow, opw,
                                         gwh, wq, wk, wvv, aowh, opwh);
  prep_bias_kernel<<<1, 256, 0, stream>>>(ipw, ipb, qn_b, kn_b, vn_b, bq, bk, bv);
  guide_gemm_kernel<<<2048, 256, 140288, stream>>>(pre, post, fused, gwh, gbuf);
  gstats_kernel<<<64, 256, 0, stream>>>(gbuf, mu, rs);
  win_attn_kernel<<<2048, 256, 265728, stream>>>(dmg, gbuf, mu, rs, gn_g, gn_b,
                                                 wq, bq, wk, bk, wvv, bv,
                                                 aowh, aob, calib);
  outproj_kernel<<<2048, 256, 108544, stream>>>(calib, fused, opwh, alpha, refined, partials);
  finalize_kernel<<<1, 64, 0, stream>>>(partials, alpha, alpha_vec, norms);
}
